// HardLSTMLayer_8340826489642
// MI455X (gfx1250) — matrix operand order repaired and checked by reading, not compiled, not run on hardware
//
#include <hip/hip_runtime.h>

#define T_STEPS 2000
#define BATCH   32
#define ISZ     512
#define HSZ     512
#define G4      2048      // 4*HSZ gate columns
#define NWG     16        // phase-2 persistent workgroups
#define SLICE   128       // gate columns per phase-2 WG (4 gates * 32 hidden)

// phase-2 LDS layout (bytes, relative to dynamic-LDS base)
#define LDS_W_BYTES (SLICE * HSZ * 2)        // 131072: w_hh slice, bf16 [128][512]
#define LDS_G_BYTES (BATCH * SLICE * 4)      //  16384: gate staging f32 [32][128]
#define LDS_H_OFF   (LDS_W_BYTES + LDS_G_BYTES)
#define LDS_H_BYTES (BATCH * HSZ * 2)        //  32768: h stage, bf16 [32][512]
#define LDS_TOTAL   (LDS_H_OFF + LDS_H_BYTES)

typedef __attribute__((ext_vector_type(16))) __bf16 v16bf;
typedef __attribute__((ext_vector_type(8)))  float  v8f;
typedef __attribute__((ext_vector_type(4)))  unsigned int u32x4;
typedef __attribute__((ext_vector_type(8)))  int    i32x8;
typedef __attribute__((ext_vector_type(4)))  int    i32x4;

union FragBF { v16bf v; uint4 u[2]; };

// A-matrix 16x32 bf16 fragment (ISA 7.12.2): lane M=L%16;
// lanes 0-15 hold K = {kb..kb+7, kb+16..kb+23}; lanes 16-31 hold K = {kb+8.., kb+24..}.
__device__ inline v16bf frag_a(const __bf16* __restrict__ base, int ld, int m0, int kb, int lane) {
  const int m = m0 + (lane & 15);
  const __bf16* p = base + (size_t)m * ld + kb + ((lane & 16) ? 8 : 0);
  FragBF r;
  r.u[0] = *(const uint4*)(p);
  r.u[1] = *(const uint4*)(p + 16);
  return r.v;
}

// B-matrix 32x16 bf16 fragment: lane N=L%16; lanes 0-15 hold K = {kb..kb+7, kb+16..kb+23},
// lanes 16-31 hold K = {kb+8.., kb+24..}. Column n contiguous in K.
__device__ inline v16bf frag_b(const __bf16* __restrict__ base, int ld, int n0, int kb, int lane) {
  const int n = n0 + (lane & 15);
  const __bf16* p = base + (size_t)n * ld + kb + ((lane & 16) ? 8 : 0);
  FragBF r;
  r.u[0] = *(const uint4*)(p);
  r.u[1] = *(const uint4*)(p + 16);
  return r.v;
}

__device__ inline float hsig(float x) { return fminf(fmaxf(0.2f * x + 0.5f, 0.f), 1.f); }
__device__ inline float htanh(float x) { return fminf(fmaxf(x, -1.f), 1.f); }

// -------------------- utility kernels --------------------

__global__ void __launch_bounds__(256)
f32_to_bf16(const float* __restrict__ src, __bf16* __restrict__ dst, int n) {
  int i = blockIdx.x * 256 + threadIdx.x;
  if (i < n) dst[i] = (__bf16)src[i];
}

__global__ void __launch_bounds__(256)
init_state(const float* __restrict__ h0, __bf16* __restrict__ hbuf, unsigned* __restrict__ bar) {
  int i = blockIdx.x * 256 + threadIdx.x;
  if (i < BATCH * HSZ) hbuf[i] = (__bf16)h0[i];
  if (blockIdx.x == 0 && threadIdx.x < 16) bar[threadIdx.x] = 0u;
}

// -------------------- phase 1: x_gates = x @ w_ih^T + b_ih + b_hh --------------------

__global__ void __launch_bounds__(256)
xgates_gemm(const __bf16* __restrict__ xb, const __bf16* __restrict__ wb,
            const float* __restrict__ b_ih, const float* __restrict__ b_hh,
            float* __restrict__ xg) {
  const int tid  = threadIdx.x;
  const int lane = tid & 31;
  const int wave = tid >> 5;                  // nt strip 0..7
  const int m0   = blockIdx.x * 32;           // row block (rows = t*B+b)
  const int n0   = blockIdx.y * 128 + wave * 16;
  const int nl   = n0 + (lane & 15);

  v8f acc0 = {0.f,0.f,0.f,0.f,0.f,0.f,0.f,0.f};
  v8f acc1 = acc0;
#pragma unroll 4
  for (int kb = 0; kb < ISZ; kb += 32) {
    v16bf b  = frag_b(wb, ISZ, n0, kb, lane);
    v16bf a0 = frag_a(xb, ISZ, m0,      kb, lane);
    v16bf a1 = frag_a(xb, ISZ, m0 + 16, kb, lane);
    acc0 = __builtin_amdgcn_wmma_f32_16x16x32_bf16(false, a0, false, b, (short)0, acc0, false, false);
    acc1 = __builtin_amdgcn_wmma_f32_16x16x32_bf16(false, a1, false, b, (short)0, acc1, false, false);
  }
  const float bias = b_ih[nl] + b_hh[nl];
  const int mrow = m0 + ((lane & 16) ? 8 : 0);
#pragma unroll
  for (int v = 0; v < 8; ++v) {
    xg[(size_t)(mrow + v)      * G4 + nl] = acc0[v] + bias;
    xg[(size_t)(mrow + 16 + v) * G4 + nl] = acc1[v] + bias;
  }
}

// -------------------- grid barrier (16 co-resident WGs) --------------------

__device__ inline void grid_barrier(unsigned* __restrict__ bar) {
  __threadfence();          // make this WG's h writes device-visible
  __syncthreads();
  if (threadIdx.x == 0) {
    unsigned gen = __atomic_load_n(bar + 1, __ATOMIC_ACQUIRE);
    unsigned arr = atomicAdd(bar, 1u);
    if (arr == NWG - 1) {
      __atomic_store_n(bar, 0u, __ATOMIC_RELAXED);
      __threadfence();
      atomicAdd(bar + 1, 1u);
    } else {
      while (__atomic_load_n(bar + 1, __ATOMIC_ACQUIRE) == gen)
        __builtin_amdgcn_s_sleep(1);
    }
    __threadfence();
  }
  __syncthreads();
}

// -------------------- TDM: DMA the 32KB contiguous h buffer into LDS ----------
// D# per CDNA5 ISA: 2D descriptor, groups 2/3 zero (<=2D tensor).
// This toolchain exposes the 6-arg builtin:
//   (u32x4 g0, i32x8 g1, i32x4 g2, i32x4 g3, i32x8 extra, i32 cpol)

__device__ inline void tdm_load_h_to_lds(const __bf16* __restrict__ hcur) {
  const unsigned long long ga = (unsigned long long)(uintptr_t)hcur;
  const unsigned lds_addr = __builtin_amdgcn_groupstaticsize() + (unsigned)LDS_H_OFF;

  u32x4 g0;
  g0[0] = 1u;                                            // count=1, is_restore=0, no gather
  g0[1] = lds_addr;                                      // lds_addr[31:0]
  g0[2] = (unsigned)(ga & 0xFFFFFFFFu);                  // global_addr[31:0]
  g0[3] = (unsigned)((ga >> 32) & 0x01FFFFFFu)           // global_addr[56:32]
          | (2u << 30);                                  // type=2 ("image")

  i32x8 g1;
  g1[0] = 0x00010000;        // workgroup_mask=0, data_size=1 (2 bytes)
  g1[1] = (int)0x80000000;   // tensor_dim0[15:0]=0x8000 (32768 elems) in bits[63:48]
  g1[2] = 0x00010000;        // tensor_dim0[31:16]=0, tensor_dim1=1
  g1[3] = (int)0x80000000;   // tile_dim0=0x8000 in bits[127:112]
  g1[4] = 0x00000001;        // tile_dim1=1, tile_dim2=0
  g1[5] = 0x00008000;        // tensor_dim0_stride=32768
  g1[6] = 0;
  g1[7] = 0;

  i32x4 z4 = {0, 0, 0, 0};
  i32x8 z8 = {0, 0, 0, 0, 0, 0, 0, 0};
  __builtin_amdgcn_tensor_load_to_lds(g0, g1, z4, z4, z8, 0);
  __builtin_amdgcn_s_wait_tensorcnt(0);
}

// -------------------- phase 2: persistent recurrent scan --------------------
// 16 WGs; WG wg owns hidden units [wg*32, wg*32+32) -> 128 gate columns.
// w_hh slice resident in LDS (128KB bf16); per step, h (32KB bf16) is DMA'd
// into LDS by the TDM, all WMMA operands then come from LDS; gates staged via
// a 32x128 f32 LDS tile; c lives in registers for all 2000 steps; h exchanged
// through a double-buffered global bf16 buffer with one grid barrier/step.

__global__ void __launch_bounds__(256)
lstm_scan(const __bf16* __restrict__ whhb, const float* __restrict__ xg,
          const float* __restrict__ c0, float* __restrict__ out,
          float* __restrict__ hT, float* __restrict__ cT,
          __bf16* __restrict__ hbuf, unsigned* __restrict__ bar) {
  extern __shared__ char smem[];
  __bf16* lds_w = (__bf16*)smem;                       // [128][512]
  float*  lds_g = (float*)(smem + LDS_W_BYTES);        // [32][128]
  __bf16* lds_h = (__bf16*)(smem + LDS_H_OFF);         // [32][512]

  const int tid   = threadIdx.x;
  const int lane  = tid & 31;
  const int wave  = tid >> 5;          // nt strip 0..7
  const int wg    = blockIdx.x;        // 0..15
  const int hbase = wg * 32;
  const int n0    = wave * 16;

  // Stage w_hh slice: LDS row r = gate*32+j  <-  w_hh row gate*512 + hbase + j.
  {
    const uint4* src = (const uint4*)whhb;   // 64 uint4 per 512-bf16 row
    uint4* dst = (uint4*)lds_w;
    for (int idx = tid; idx < SLICE * 64; idx += 256) {
      int r = idx >> 6, c = idx & 63;
      int gate = r >> 5, j = r & 31;
      dst[idx] = src[(size_t)(gate * HSZ + hbase + j) * 64 + c];
    }
  }

  // Each thread owns 4 (batch, hidden) cells; c kept in registers.
  float creg[4];
  int bq[4], jq[4];
#pragma unroll
  for (int q = 0; q < 4; ++q) {
    int idx = tid * 4 + q;
    bq[q] = idx >> 5;  jq[q] = idx & 31;
    creg[q] = c0[(size_t)bq[q] * HSZ + hbase + jq[q]];
  }
  __syncthreads();

  int phase = 0;
  for (int t = 0; t < T_STEPS; ++t) {
    const float* xgt = xg + (size_t)t * BATCH * G4;

    // DMA h into LDS (one wave issues; TDM ignores EXEC), then publish to WG.
    if (wave == 0) {
      tdm_load_h_to_lds(hbuf + (size_t)phase * BATCH * HSZ);
    }
    // Prefetch this step's x_gate rows while waiting / doing the GEMM.
#pragma unroll
    for (int q = 0; q < 4; ++q)
      __builtin_prefetch(xgt + (size_t)bq[q] * G4 + hbase + jq[q], 0, 1);
    __syncthreads();

    v8f acc0 = {0.f,0.f,0.f,0.f,0.f,0.f,0.f,0.f};
    v8f acc1 = acc0;
#pragma unroll 4
    for (int kb = 0; kb < HSZ; kb += 32) {
      v16bf b  = frag_b(lds_w, HSZ, n0, kb, lane);   // ds_load_b128
      v16bf a0 = frag_a(lds_h, HSZ, 0,  kb, lane);   // ds_load_b128
      v16bf a1 = frag_a(lds_h, HSZ, 16, kb, lane);
      acc0 = __builtin_amdgcn_wmma_f32_16x16x32_bf16(false, a0, false, b, (short)0, acc0, false, false);
      acc1 = __builtin_amdgcn_wmma_f32_16x16x32_bf16(false, a1, false, b, (short)0, acc1, false, false);
    }

    // Stage gate pre-activations to LDS [32 batch][128 cols].
    const int nl = n0 + (lane & 15);
    const int mrow = (lane & 16) ? 8 : 0;
#pragma unroll
    for (int v = 0; v < 8; ++v) {
      lds_g[(mrow + v)      * SLICE + nl] = acc0[v];
      lds_g[(mrow + 16 + v) * SLICE + nl] = acc1[v];
    }
    __syncthreads();

    __bf16* hnext = hbuf + (size_t)(phase ^ 1) * BATCH * HSZ;
#pragma unroll
    for (int q = 0; q < 4; ++q) {
      const int b_ = bq[q], j = jq[q];
      const float* xr = xgt + (size_t)b_ * G4 + hbase + j;
      float gi = lds_g[b_ * SLICE + j]       + xr[0];
      float gf = lds_g[b_ * SLICE + 32 + j]  + xr[512];
      float gg = lds_g[b_ * SLICE + 64 + j]  + xr[1024];
      float go = lds_g[b_ * SLICE + 96 + j]  + xr[1536];
      float i_ = hsig(gi), f_ = hsig(gf), g_ = htanh(gg), o_ = hsig(go);
      float c  = f_ * creg[q] + i_ * g_;
      creg[q]  = c;
      float h  = o_ * htanh(c);
      out[((size_t)t * BATCH + b_) * HSZ + hbase + j] = h;
      hnext[(size_t)b_ * HSZ + hbase + j] = (__bf16)h;
      if (t == T_STEPS - 1) hT[(size_t)b_ * HSZ + hbase + j] = h;
    }
    phase ^= 1;
    grid_barrier(bar);
  }
#pragma unroll
  for (int q = 0; q < 4; ++q)
    cT[(size_t)bq[q] * HSZ + hbase + jq[q]] = creg[q];
}

// -------------------- launch --------------------

extern "C" void kernel_launch(void* const* d_in, const int* in_sizes, int n_in,
                              void* d_out, int out_size, void* d_ws, size_t ws_size,
                              hipStream_t stream) {
  const float* x    = (const float*)d_in[0];
  const float* h0   = (const float*)d_in[1];
  const float* c0   = (const float*)d_in[2];
  const float* w_ih = (const float*)d_in[3];
  const float* w_hh = (const float*)d_in[4];
  const float* b_ih = (const float*)d_in[5];
  const float* b_hh = (const float*)d_in[6];

  // Workspace layout (256B-aligned):
  char* ws = (char*)d_ws;
  __bf16*   xb   = (__bf16*)(ws);                  // T*B*I bf16      = 65,536,000 B
  __bf16*   wihb = (__bf16*)(ws + 65536000);       // 4H*I bf16       =  2,097,152 B
  __bf16*   whhb = (__bf16*)(ws + 67633152);       // 4H*H bf16       =  2,097,152 B
  float*    xg   = (float*) (ws + 69730304);       // T*B*4H f32      = 524,288,000 B
  __bf16*   hbuf = (__bf16*)(ws + 594018304);      // 2 * B*H bf16    =     65,536 B
  unsigned* bar  = (unsigned*)(ws + 594083840);    // barrier state

  float* out = (float*)d_out;
  float* hT  = out + (size_t)T_STEPS * BATCH * HSZ;
  float* cT  = hT + (size_t)BATCH * HSZ;

  // bf16 conversions
  const int nx = T_STEPS * BATCH * ISZ;
  const int nw = G4 * ISZ;
  f32_to_bf16<<<(nx + 255) / 256, 256, 0, stream>>>(x, xb, nx);
  f32_to_bf16<<<(nw + 255) / 256, 256, 0, stream>>>(w_ih, wihb, nw);
  f32_to_bf16<<<(nw + 255) / 256, 256, 0, stream>>>(w_hh, whhb, nw);
  init_state<<<(BATCH * HSZ + 255) / 256, 256, 0, stream>>>(h0, hbuf, bar);

  // Phase 1: big WMMA GEMM for x_gates
  xgates_gemm<<<dim3(T_STEPS, 16), 256, 0, stream>>>(xb, wihb, b_ih, b_hh, xg);

  // Phase 2: persistent recurrent scan
  // LDS: 128KB w_hh slice + 16KB gate tile + 32KB h stage = 176KB / WGP
  const int smem_bytes = LDS_TOTAL;
  (void)hipFuncSetAttribute((const void*)lstm_scan,
                            hipFuncAttributeMaxDynamicSharedMemorySize, smem_bytes);
  lstm_scan<<<NWG, 256, smem_bytes, stream>>>(whhb, xg, c0, out, hT, cT, hbuf, bar);
}
